// Attention_70257075028315
// MI455X (gfx1250) — matrix operand order repaired and checked by reading, not compiled, not run on hardware
//
#include <hip/hip_runtime.h>
#include <hip/hip_bf16.h>
#include <math.h>

// ---------------------------------------------------------------------------
// Attention (B=8, Nq=Nk=2048, D=512), V == K, unscaled softmax.
// Flash-attention style, bf16 WMMA (v_wmma_f32_16x16x32_bf16), fp32 accum.
//  - All WMMA operand fragments are contiguous 32-byte LDS loads (swizzled Q/P
//    layouts, dual-layout K tile: row-major Ks + transposed KsT).
//  - K tile double-buffered in LDS via GLOBAL_LOAD_ASYNC_TO_LDS_B128
//    (ASYNCcnt pipeline), sync fallback if builtin is absent.
//  - Softmax pass parallelized across all 8 waves.
// ---------------------------------------------------------------------------

typedef __attribute__((ext_vector_type(16))) __bf16 v16bf;
typedef __attribute__((ext_vector_type(8)))  float  v8f;
typedef int v4i_vec __attribute__((vector_size(16)));

#define B_SZ    8
#define NQ      2048
#define NK      2048
#define DH      512
#define MT      32      // query rows per workgroup
#define NT      64      // key tile per iteration
#define THREADS 256     // 8 waves (wave32)

// LDS layout (bytes):
//   Qs  : MT*DH bf16 (swizzled)   = 32768   @ 0
//   Ks0 : NT*DH bf16 (row-major)  = 65536   @ 32768
//   Ks1 : NT*DH bf16 (row-major)  = 65536   @ 98304   (async double buffer)
//   KsT : DH*NT bf16 (transposed) = 65536   @ 163840
//   Ss  : MT*NT f32               = 8192    @ 229376
//   Ps  : MT*NT bf16 (swizzled)   = 4096    @ 237568
//   mS/lS/cS : 3*MT f32           = 384     @ 241664
//   pbuf: 256 f32 partials        = 1024    @ 242048
#define SMEM_BYTES 243072

#if defined(__has_builtin)
#  if __has_builtin(__builtin_amdgcn_global_load_async_to_lds_b128)
#    define HAVE_ASYNC_LDS 1
#  endif
#endif
#ifndef HAVE_ASYNC_LDS
#  define HAVE_ASYNC_LDS 0
#endif

#if HAVE_ASYNC_LDS
#  if __has_builtin(__builtin_amdgcn_s_wait_asynccnt)
#    define WAIT_ASYNC() __builtin_amdgcn_s_wait_asynccnt(0)
#  else
#    define WAIT_ASYNC() asm volatile("s_wait_asynccnt 0" ::: "memory")
#  endif
#endif

union FragBF {
  v16bf v;
  uint4 q;
  __bf16 h[16];
  unsigned int u[8];
};

union Vec8H {
  uint4 q;
  unsigned short s[8];
  __bf16 h[8];
};

__device__ __constant__ int kSlot[4] = {0, 1, 2, 3};  // identity (contiguous k order)

__global__ void __launch_bounds__(256)
cvt_bf16_kernel(const float* __restrict__ x, __bf16* __restrict__ y, int n) {
  int i = blockIdx.x * blockDim.x + threadIdx.x;
  int stride = gridDim.x * blockDim.x;
  for (; i < n; i += stride) y[i] = (__bf16)x[i];
}

#if HAVE_ASYNC_LDS
// AS(1) = global ("__device__"), AS(3) = LDS ("__shared__") int4 pointees.
typedef __attribute__((address_space(1))) v4i_vec* async_gptr;
typedef __attribute__((address_space(3))) v4i_vec* async_lptr;

__device__ __forceinline__ void async_copy16(const __bf16* g, __bf16* l) {
  __builtin_amdgcn_global_load_async_to_lds_b128(
      (async_gptr)(void*)g, (async_lptr)(void*)l, 0, 0);
}

// One NT x DH tile, 16B per async op, 16 ops per thread.
__device__ __forceinline__ void fill_tile_async(const __bf16* __restrict__ src,
                                                __bf16* dst, int tid) {
  #pragma unroll
  for (int i = 0; i < 16; ++i) {
    const int idx = tid + i * THREADS;      // 0..4095
    const int key = idx >> 6;
    const int d0  = (idx & 63) * 8;
    async_copy16(src + (size_t)key * DH + d0, dst + key * DH + d0);
  }
}
#endif

__global__ void __launch_bounds__(THREADS)
attn_fa_kernel(const __bf16* __restrict__ Qg, const __bf16* __restrict__ Kg,
               float* __restrict__ Og) {
  extern __shared__ char smem[];
  __bf16* Qs  = (__bf16*)(smem);
  __bf16* Ks0 = (__bf16*)(smem + 32768);
  __bf16* Ks1 = (__bf16*)(smem + 98304);
  __bf16* KsT = (__bf16*)(smem + 163840);
  float*  Ss  = (float*) (smem + 229376);
  __bf16* Ps  = (__bf16*)(smem + 237568);
  float*  mS  = (float*) (smem + 241664);
  float*  lS  = mS + MT;
  float*  cS  = lS + MT;
  float*  pbuf= (float*) (smem + 242048);

  const int tid  = threadIdx.x;
  const int wave = tid >> 5;
  const int lane = tid & 31;
  const int l16  = lane & 15;
  const int hi   = lane >> 4;          // 0 for lanes 0-15, 1 for 16-31

  const int qt = blockIdx.x % (NQ / MT);
  const int b  = blockIdx.x / (NQ / MT);
  const int q0 = qt * MT;

  const __bf16* KgB = Kg + (size_t)b * NK * DH;

#if HAVE_ASYNC_LDS
  // kick off async fill of tile 0 before anything else
  fill_tile_async(KgB, Ks0, tid);
#endif

  // ---- stage Q tile into LDS, swizzled per 32-chunk ----
  {
    const uint4* src = (const uint4*)(Qg + ((size_t)b * NQ + q0) * DH);
    const int nvec = MT * DH / 8;      // 2048 x 16B
    for (int i = tid; i < nvec; i += THREADS) {
      const int m  = i >> 6;           // row
      const int k  = (i & 63) * 8;
      const int chunk = k >> 5;
      const int g     = (k & 31) >> 3;
      uint4 v = src[i];
      *(uint4*)(Qs + m * DH + chunk * 32 + kSlot[g] * 8) = v;
    }
  }
  if (tid < MT) { mS[tid] = -1e30f; lS[tid] = 0.0f; }

  // stage-1 wave mapping: one 16x16 S tile per wave
  const int qsub1 = wave >> 2;                 // 0..1 query subtile
  const int n0s   = (wave & 3) * 16;           // score-column base
  const int mrowS = qsub1 * 16 + l16;          // A row (Q row)

  // stage-2 wave mapping: 2 query subtiles x 4 column groups of 128
  const int qsub2 = wave & 1;
  const int c0    = (wave >> 1) * 128;

  // softmax-pass mapping: 256 threads = 32 rows x 8 col-octets
  const int smR = tid >> 3;
  const int smG = tid & 7;

  v8f oacc[8];
  #pragma unroll
  for (int t = 0; t < 8; ++t) oacc[t] = (v8f){};

  for (int kt = 0; kt < NK / NT; ++kt) {
    const int k0 = kt * NT;

#if HAVE_ASYNC_LDS
    __bf16* Kcur = (kt & 1) ? Ks1 : Ks0;
    __bf16* Knxt = (kt & 1) ? Ks0 : Ks1;
    WAIT_ASYNC();        // our share of tile kt is in LDS
    __syncthreads();     // all waves' shares in; prev consumers of Knxt done
    if (k0 + NT < NK) fill_tile_async(KgB + (size_t)(k0 + NT) * DH, Knxt, tid);

    // build transposed copy KsT[d][key] from Kcur (LDS->regs->LDS)
    {
      #pragma unroll
      for (int i = 0; i < 8; ++i) {
        const int task = tid + i * THREADS;
        const int d0  = (task & 63) * 8;
        const int key = (task >> 6) * 2;
        Vec8H a, bb;
        a.q  = *(const uint4*)(Kcur + key * DH + d0);
        bb.q = *(const uint4*)(Kcur + (key + 1) * DH + d0);
        #pragma unroll
        for (int j = 0; j < 8; ++j) {
          const unsigned int pk =
              (unsigned int)a.s[j] | ((unsigned int)bb.s[j] << 16);
          *(unsigned int*)(KsT + (d0 + j) * NT + key) = pk;
        }
      }
    }
#else
    __bf16* Kcur = Ks0;
    __syncthreads();     // previous iteration done with Ks/KsT/Ps/stats
    // synchronous staging: global -> (Ks row-major, KsT transposed)
    {
      const __bf16* src = KgB + (size_t)k0 * DH;
      #pragma unroll
      for (int i = 0; i < 8; ++i) {
        const int task = tid + i * THREADS;
        const int d0  = (task & 63) * 8;
        const int key = (task >> 6) * 2;
        Vec8H a, bb;
        a.q  = *(const uint4*)(src + (size_t)key * DH + d0);
        bb.q = *(const uint4*)(src + (size_t)(key + 1) * DH + d0);
        *(uint4*)(Kcur + key * DH + d0)       = a.q;
        *(uint4*)(Kcur + (key + 1) * DH + d0) = bb.q;
        #pragma unroll
        for (int j = 0; j < 8; ++j) {
          const unsigned int pk =
              (unsigned int)a.s[j] | ((unsigned int)bb.s[j] << 16);
          *(unsigned int*)(KsT + (d0 + j) * NT + key) = pk;
        }
      }
    }
    if (k0 + NT < NK) {
      const char* nxt = (const char*)(KgB + (size_t)(k0 + NT) * DH);
      __builtin_prefetch(nxt + tid * 256, 0, 1);
    }
#endif
    __syncthreads();

    // ---- stage 1: S = Q * K^T, one 16x16 tile per wave, K-dim = 512 ----
    {
      v8f acc = (v8f){};
      const int aOff   = hi ? 16 : 0;    // swizzled A: hi lanes read pos 16..31
      const int keyRow = n0s + l16;      // B column -> key index in tile
      const int dbB    = hi ? 16 : 0;    // B K-dim base
      #pragma unroll
      for (int kc = 0; kc < DH / 32; ++kc) {
        const int dc = kc * 32;
        FragBF a, bb;
        a.v  = *(const v16bf*)(Qs + mrowS * DH + dc + aOff);
        bb.v = *(const v16bf*)(Kcur + keyRow * DH + dc + dbB);
        acc = __builtin_amdgcn_wmma_f32_16x16x32_bf16(
            false, a.v, false, bb.v, (short)0, acc, false, false);
      }
      const int scol = n0s + l16;
      #pragma unroll
      for (int v = 0; v < 8; ++v)
        Ss[(qsub1 * 16 + v + hi * 8) * NT + scol] = acc[v];
    }
    __syncthreads();

    // ---- softmax pass A: per-thread partial max over 8 columns ----
    {
      const float4* Srow = (const float4*)(Ss + smR * NT + smG * 8);
      float4 s4a = Srow[0], s4b = Srow[1];
      float tm = fmaxf(fmaxf(fmaxf(s4a.x, s4a.y), fmaxf(s4a.z, s4a.w)),
                       fmaxf(fmaxf(s4b.x, s4b.y), fmaxf(s4b.z, s4b.w)));
      pbuf[tid] = tm;
    }
    __syncthreads();

    // ---- softmax pass B: per-row combine, update m, compute correction ----
    if (tid < MT) {
      const int r = tid;
      float tm = pbuf[r * 8];
      #pragma unroll
      for (int j = 1; j < 8; ++j) tm = fmaxf(tm, pbuf[r * 8 + j]);
      const float mOld = mS[r];
      const float mNew = fmaxf(mOld, tm);
      cS[r] = __expf(mOld - mNew);
      mS[r] = mNew;
    }
    __syncthreads();

    // ---- softmax pass C: exp, bf16 P (swizzled), partial sums ----
    {
      const float mNew = mS[smR];
      const float4* Srow = (const float4*)(Ss + smR * NT + smG * 8);
      float4 s4a = Srow[0], s4b = Srow[1];
      Vec8H p8;
      float e0 = __expf(s4a.x - mNew), e1 = __expf(s4a.y - mNew);
      float e2 = __expf(s4a.z - mNew), e3 = __expf(s4a.w - mNew);
      float e4 = __expf(s4b.x - mNew), e5 = __expf(s4b.y - mNew);
      float e6 = __expf(s4b.z - mNew), e7 = __expf(s4b.w - mNew);
      p8.h[0] = (__bf16)e0; p8.h[1] = (__bf16)e1;
      p8.h[2] = (__bf16)e2; p8.h[3] = (__bf16)e3;
      p8.h[4] = (__bf16)e4; p8.h[5] = (__bf16)e5;
      p8.h[6] = (__bf16)e6; p8.h[7] = (__bf16)e7;
      const int chunk = smG >> 2;
      const int gg    = smG & 3;
      *(uint4*)(Ps + smR * NT + chunk * 32 + kSlot[gg] * 8) = p8.q;
      pbuf[tid] = e0 + e1 + e2 + e3 + e4 + e5 + e6 + e7;
    }
    __syncthreads();

    // ---- softmax pass D: per-row l update ----
    if (tid < MT) {
      const int r = tid;
      float s = pbuf[r * 8];
      #pragma unroll
      for (int j = 1; j < 8; ++j) s += pbuf[r * 8 + j];
      lS[r] = lS[r] * cS[r] + s;
    }

    // ---- stage 2b: rescale O, then O += P * V (V = K tile) ----
    {
      float corr_v[8];
      #pragma unroll
      for (int v = 0; v < 8; ++v) corr_v[v] = cS[qsub2 * 16 + v + hi * 8];
      #pragma unroll
      for (int t = 0; t < 8; ++t)
        #pragma unroll
        for (int v = 0; v < 8; ++v) oacc[t][v] *= corr_v[v];

      const int mrowP = qsub2 * 16 + l16;
      const int aOff  = hi ? 16 : 0;     // swizzled P A-fragment
      const int kbB   = hi ? 16 : 0;     // PV B K-dim (key) base
      #pragma unroll
      for (int t = 0; t < 8; ++t) {
        const int ncol = c0 + t * 16 + l16;   // output D column for this lane
        #pragma unroll
        for (int kc = 0; kc < NT / 32; ++kc) {
          const int dc = kc * 32;
          FragBF a, bb;
          a.v  = *(const v16bf*)(Ps + mrowP * NT + dc + aOff);
          bb.v = *(const v16bf*)(KsT + ncol * NT + dc + kbB);
          oacc[t] = __builtin_amdgcn_wmma_f32_16x16x32_bf16(
              false, a.v, false, bb.v, (short)0, oacc[t], false, false);
        }
      }
    }
  }
  __syncthreads();

  // ---- epilogue: O /= l, store fp32 ----
  {
    float rl[8];
    #pragma unroll
    for (int v = 0; v < 8; ++v) rl[v] = 1.0f / lS[qsub2 * 16 + v + hi * 8];
    #pragma unroll
    for (int t = 0; t < 8; ++t) {
      const int col = c0 + t * 16 + l16;
      #pragma unroll
      for (int v = 0; v < 8; ++v) {
        const int row = q0 + qsub2 * 16 + v + hi * 8;
        Og[((size_t)b * NQ + row) * DH + col] = oacc[t][v] * rl[v];
      }
    }
  }
}

extern "C" void kernel_launch(void* const* d_in, const int* in_sizes, int n_in,
                              void* d_out, int out_size, void* d_ws, size_t ws_size,
                              hipStream_t stream) {
  const float* q = (const float*)d_in[0];
  const float* k = (const float*)d_in[1];
  float* out = (float*)d_out;

  const size_t N = (size_t)B_SZ * NQ * DH;   // 8.39M elements per tensor
  __bf16* qb = (__bf16*)d_ws;
  __bf16* kb = qb + N;

  cvt_bf16_kernel<<<1024, 256, 0, stream>>>(q, qb, (int)N);
  cvt_bf16_kernel<<<1024, 256, 0, stream>>>(k, kb, (int)N);

  (void)hipFuncSetAttribute((const void*)attn_fa_kernel,
                            hipFuncAttributeMaxDynamicSharedMemorySize,
                            (int)SMEM_BYTES);

  dim3 grid(B_SZ * (NQ / MT));   // 512 workgroups
  dim3 block(THREADS);
  attn_fa_kernel<<<grid, block, SMEM_BYTES, stream>>>(qb, kb, out);
}
